// BiPCN_36653250904668
// MI455X (gfx1250) — matrix operand order repaired and checked by reading, not compiled, not run on hardware
//
#include <hip/hip_runtime.h>

typedef _Float16 f16;
typedef __attribute__((ext_vector_type(4)))  _Float16 v4h;
typedef __attribute__((ext_vector_type(8)))  _Float16 v8h;
typedef __attribute__((ext_vector_type(16))) _Float16 v16h;
typedef __attribute__((ext_vector_type(4)))  float    v4f;
typedef __attribute__((ext_vector_type(8)))  float    v8f;
typedef __attribute__((ext_vector_type(4)))  int      v4i;

#define LDS_STRIDE 40   // 32 K-elems + 8 f16 pad: 80B row stride, 16B-aligned, conflict-free

// CDNA5 async global->LDS path (ASYNCcnt), guarded so compile never fails.
#if defined(__AMDGCN__) && __has_builtin(__builtin_amdgcn_global_load_async_to_lds_b128) && \
    __has_builtin(__builtin_amdgcn_s_wait_asynccnt)
#define USE_ASYNC_LDS 1
#else
#define USE_ASYNC_LDS 0
#endif

union V16H { v16h v; v8h h[2]; };

#if USE_ASYNC_LDS
// b128 async copy: param 0 = global (__device__/AS1) v4i*, param 1 = LDS (AS3) v4i*,
// then imm offset, imm cpol (arity/type confirmed by compiler diagnostics).
__device__ __forceinline__ void stage16_async(const f16* __restrict__ g, f16* l) {
  __builtin_amdgcn_global_load_async_to_lds_b128(
      (__attribute__((address_space(1))) v4i*)(g),
      (__attribute__((address_space(3))) v4i*)(l), 0, 0);
}
#endif

// ---------------- elementwise f32 -> f16 convert ----------------
__global__ __launch_bounds__(256) void cvt_f32_f16(const float* __restrict__ in,
                                                   f16* __restrict__ out, int n4) {
  int i = blockIdx.x * 256 + threadIdx.x;
  if (i >= n4) return;
  v4f v = ((const v4f*)in)[i];
  v4h h;
  h.x = (f16)v.x; h.y = (f16)v.y; h.z = (f16)v.z; h.w = (f16)v.w;
  ((v4h*)out)[i] = h;
}

// ---------------- tiled transpose f32 -> f16 : out[c*R+r] = in[r*C+c] ----------------
__global__ __launch_bounds__(256) void transpose_f32_f16(const float* __restrict__ in,
                                                         f16* __restrict__ out,
                                                         int R, int C) {
  __shared__ float tile[32][33];
  int bx = blockIdx.x * 32;      // col base in `in`
  int by = blockIdx.y * 32;      // row base in `in`
  int tx = threadIdx.x;          // 0..31
  int ty = threadIdx.y;          // 0..7
#pragma unroll
  for (int i = 0; i < 32; i += 8)
    tile[ty + i][tx] = in[(size_t)(by + ty + i) * C + (bx + tx)];
  __syncthreads();
#pragma unroll
  for (int i = 0; i < 32; i += 8)
    out[(size_t)(bx + ty + i) * R + (by + tx)] = (f16)tile[tx][ty + i];
}

// ---------------- fused dual-GEMM with pointwise epilogue ----------------
// out = alpha1*(A1@B1) + alpha2*(A2@B2) + sCoef*S + b1*P1 + b2*P2
// A: MxK f16 row-major.  B is passed PRE-TRANSPOSED as Bt: NxK f16 row-major,
// so both A and B tiles stage into LDS with contiguous 16B chunks.
// M,N multiples of 128; K multiples of 32.
__global__ __launch_bounds__(256) void gemm_fused(
    const f16* __restrict__ A1, const f16* __restrict__ Bt1, int K1,
    const f16* __restrict__ A2, const f16* __restrict__ Bt2, int K2,
    float alpha1, float alpha2,
    const float* __restrict__ S, float sCoef,
    const f16* __restrict__ P1, float b1,
    const f16* __restrict__ P2, float b2,
    float* __restrict__ outF, f16* __restrict__ outH,
    int M, int N)
{
  __shared__ f16 ldsA[2][128 * LDS_STRIDE];   // [m][k], double buffered
  __shared__ f16 ldsB[2][128 * LDS_STRIDE];   // [n][k], double buffered

  const int t    = threadIdx.x;
  const int lane = t & 31;
  const int wave = t >> 5;           // 8 wave32s
  const int wr   = (wave >> 2) * 64; // wave M-base inside block tile
  const int wc   = (wave & 3) * 32;  // wave N-base inside block tile
  const int half = lane >> 4;        // ISA half-wave select
  const int lrow = lane & 15;

  const int mBase = blockIdx.y * 128;
  const int nBase = blockIdx.x * 128;

  const v8f vzero = {0.f,0.f,0.f,0.f,0.f,0.f,0.f,0.f};
  v8f acc1[4][2], acc2[4][2];
#pragma unroll
  for (int i = 0; i < 4; ++i)
#pragma unroll
    for (int j = 0; j < 2; ++j) { acc1[i][j] = vzero; acc2[i][j] = vzero; }

  // tile-loader mapping: 256 threads x 2 chunks of 8 f16 cover 128x32
  const int ar = t >> 2;             // row 0..63 (second pass +64)
  const int ac = (t & 3) * 8;        // col chunk {0,8,16,24}

  auto compute = [&](int cur, v8f (&acc)[4][2]) {
    // fragments per CDNA5 16-bit WMMA layouts: two ds_load_b128 each
    V16H af[4], bf[2];
#pragma unroll
    for (int mt = 0; mt < 4; ++mt) {
      int row = wr + mt * 16 + lrow;
      af[mt].h[0] = *(const v8h*)&ldsA[cur][row * LDS_STRIDE + half * 8];
      af[mt].h[1] = *(const v8h*)&ldsA[cur][row * LDS_STRIDE + 16 + half * 8];
    }
#pragma unroll
    for (int nt = 0; nt < 2; ++nt) {
      int col = wc + nt * 16 + lrow;
      bf[nt].h[0] = *(const v8h*)&ldsB[cur][col * LDS_STRIDE + half * 8];
      bf[nt].h[1] = *(const v8h*)&ldsB[cur][col * LDS_STRIDE + 16 + half * 8];
    }
#pragma unroll
    for (int mt = 0; mt < 4; ++mt)
#pragma unroll
      for (int nt = 0; nt < 2; ++nt)
        acc[mt][nt] = __builtin_amdgcn_wmma_f32_16x16x32_f16(
            false, af[mt].v, false, bf[nt].v, (short)0, acc[mt][nt], false, false);
  };

  auto run = [&](const f16* __restrict__ Am, const f16* __restrict__ Bt, int K,
                 v8f (&acc)[4][2]) {
    int cur = 0;
#if USE_ASYNC_LDS
    // prologue: async-stage tile 0
#pragma unroll
    for (int p = 0; p < 2; ++p) {
      int r = ar + p * 64;
      stage16_async(&Am[(size_t)(mBase + r) * K + ac], &ldsA[0][r * LDS_STRIDE + ac]);
      stage16_async(&Bt[(size_t)(nBase + r) * K + ac], &ldsB[0][r * LDS_STRIDE + ac]);
    }
    for (int kt = 0; kt < K; kt += 32) {
      __builtin_amdgcn_s_wait_asynccnt(0);   // own async fills for tile kt done
      __syncthreads();                       // everyone's done -> buffer cur valid
      if (kt + 32 < K) {                     // async-prefetch next tile into other buffer
#pragma unroll
        for (int p = 0; p < 2; ++p) {
          int r = ar + p * 64;
          stage16_async(&Am[(size_t)(mBase + r) * K + kt + 32 + ac],
                        &ldsA[cur ^ 1][r * LDS_STRIDE + ac]);
          stage16_async(&Bt[(size_t)(nBase + r) * K + kt + 32 + ac],
                        &ldsB[cur ^ 1][r * LDS_STRIDE + ac]);
        }
      }
      compute(cur, acc);
      cur ^= 1;
    }
#else
    // register-pipelined sync path: global loads overlap WMMA, ds_store after compute
    v8h pa[2], pb[2];
#pragma unroll
    for (int p = 0; p < 2; ++p) {
      int r = ar + p * 64;
      pa[p] = *(const v8h*)&Am[(size_t)(mBase + r) * K + ac];
      pb[p] = *(const v8h*)&Bt[(size_t)(nBase + r) * K + ac];
    }
#pragma unroll
    for (int p = 0; p < 2; ++p) {
      int r = ar + p * 64;
      *(v8h*)&ldsA[0][r * LDS_STRIDE + ac] = pa[p];
      *(v8h*)&ldsB[0][r * LDS_STRIDE + ac] = pb[p];
    }
    for (int kt = 0; kt < K; kt += 32) {
      __syncthreads();                       // buffer cur valid
      const bool more = (kt + 32) < K;
      if (more) {                            // issue next-tile global loads early
#pragma unroll
        for (int p = 0; p < 2; ++p) {
          int r = ar + p * 64;
          pa[p] = *(const v8h*)&Am[(size_t)(mBase + r) * K + kt + 32 + ac];
          pb[p] = *(const v8h*)&Bt[(size_t)(nBase + r) * K + kt + 32 + ac];
        }
      }
      compute(cur, acc);
      if (more) {                            // land prefetch into other buffer
#pragma unroll
        for (int p = 0; p < 2; ++p) {
          int r = ar + p * 64;
          *(v8h*)&ldsA[cur ^ 1][r * LDS_STRIDE + ac] = pa[p];
          *(v8h*)&ldsB[cur ^ 1][r * LDS_STRIDE + ac] = pb[p];
        }
      }
      cur ^= 1;
    }
#endif
    __syncthreads();   // protect LDS before next phase / exit
  };

  run(A1, Bt1, K1, acc1);
  if (K2 > 0) run(A2, Bt2, K2, acc2);

  // epilogue: D layout -> lane holds col (lrow), rows half*8 + 0..7
#pragma unroll
  for (int mt = 0; mt < 4; ++mt) {
#pragma unroll
    for (int nt = 0; nt < 2; ++nt) {
      int rowBase = mBase + wr + mt * 16 + half * 8;
      int col     = nBase + wc + nt * 16 + lrow;
#pragma unroll
      for (int r = 0; r < 8; ++r) {
        size_t idx = (size_t)(rowBase + r) * N + col;
        float v = alpha1 * acc1[mt][nt][r] + alpha2 * acc2[mt][nt][r];
        if (S)    v += sCoef * S[idx];
        if (P1)   v += b1 * (float)P1[idx];
        if (P2)   v += b2 * (float)P2[idx];
        if (outF) outF[idx] = v;
        if (outH) outH[idx] = (f16)v;
      }
    }
  }
}

// ---------------- host orchestration ----------------
extern "C" void kernel_launch(void* const* d_in, const int* in_sizes, int n_in,
                              void* d_out, int out_size, void* d_ws, size_t ws_size,
                              hipStream_t stream) {
  (void)in_sizes; (void)n_in; (void)out_size; (void)ws_size;
  const int Bsz = 4096, D0 = 1024, D1 = 2048;
  const float* x  = (const float*)d_in[0];
  const float* V0 = (const float*)d_in[1];
  const float* W0 = (const float*)d_in[2];
  const float* V1 = (const float*)d_in[3];
  const float* W1 = (const float*)d_in[4];
  const float* V2 = (const float*)d_in[5];
  const float* W2 = (const float*)d_in[6];

  char* ws = (char*)d_ws;
  size_t off = 0;
  auto alloc = [&](size_t bytes) {
    void* p = ws + off; off += (bytes + 255) & ~(size_t)255; return p;
  };

  // f16 weights in both layouts: plain (row-major) and transposed.
  // GEMM consumes B as NxK, so "X@W^T" uses Wh and "X@W" uses Wt.
  f16* xh  = (f16*)alloc((size_t)Bsz*D0*2);
  f16* V1h = (f16*)alloc((size_t)D1*D1*2);
  f16* V2h = (f16*)alloc((size_t)D1*D0*2);
  f16* W0h = (f16*)alloc((size_t)D1*D0*2);
  f16* W1h = (f16*)alloc((size_t)D1*D1*2);
  f16* W2h = (f16*)alloc((size_t)D0*D1*2);
  f16* V0t = (f16*)alloc((size_t)D1*D0*2);   // V0^T (2048x1024)
  f16* V1t = (f16*)alloc((size_t)D1*D1*2);   // V1^T (2048x2048)
  f16* V2t = (f16*)alloc((size_t)D0*D1*2);   // V2^T (1024x2048)
  f16* W0t = (f16*)alloc((size_t)D0*D1*2);   // W0^T (1024x2048)
  f16* W1t = (f16*)alloc((size_t)D1*D1*2);   // W1^T (2048x2048)
  f16* W2t = (f16*)alloc((size_t)D1*D0*2);   // W2^T (2048x1024)
  float* s1 = (float*)alloc((size_t)Bsz*D1*4);
  float* s2 = (float*)alloc((size_t)Bsz*D1*4);
  float* s3 = (float*)alloc((size_t)Bsz*D0*4);
  f16* s1h = (f16*)alloc((size_t)Bsz*D1*2);
  f16* s2h = (f16*)alloc((size_t)Bsz*D1*2);
  f16* s3h = (f16*)alloc((size_t)Bsz*D0*2);
  f16* u0  = (f16*)alloc((size_t)Bsz*D1*2);
  f16* u1  = (f16*)alloc((size_t)Bsz*D1*2);
  f16* u2  = (f16*)alloc((size_t)Bsz*D0*2);
  f16* d0  = (f16*)alloc((size_t)Bsz*D0*2);
  f16* d1  = (f16*)alloc((size_t)Bsz*D1*2);
  f16* d2  = (f16*)alloc((size_t)Bsz*D1*2);

  auto cvt = [&](const float* in, f16* out, size_t n) {
    int n4 = (int)(n / 4);
    cvt_f32_f16<<<dim3((n4 + 255)/256), dim3(256), 0, stream>>>(in, out, n4);
  };
  auto tpose = [&](const float* in, f16* out, int R, int C) {
    transpose_f32_f16<<<dim3(C/32, R/32), dim3(32,8), 0, stream>>>(in, out, R, C);
  };
  auto gemm = [&](const f16* A1, const f16* Bt1, int K1,
                  const f16* A2, const f16* Bt2, int K2,
                  float a1, float a2, const float* S, float sc,
                  const f16* P1, float c1, const f16* P2, float c2,
                  float* oF, f16* oH, int M, int N) {
    gemm_fused<<<dim3(N/128, M/128), dim3(256), 0, stream>>>(
        A1, Bt1, K1, A2, Bt2, K2, a1, a2, S, sc, P1, c1, P2, c2, oF, oH, M, N);
  };

  // one-time f16 copies / transposes
  cvt(x,  xh,  (size_t)Bsz*D0);
  cvt(V1, V1h, (size_t)D1*D1);
  cvt(V2, V2h, (size_t)D1*D0);
  cvt(W0, W0h, (size_t)D1*D0);
  cvt(W1, W1h, (size_t)D1*D1);
  cvt(W2, W2h, (size_t)D0*D1);
  tpose(V0, V0t, D0, D1);
  tpose(V1, V1t, D1, D1);
  tpose(V2, V2t, D1, D0);
  tpose(W0, W0t, D1, D0);
  tpose(W1, W1t, D1, D1);
  tpose(W2, W2t, D0, D1);

  // feed-forward init: s_{l+1} = s_l @ V_l   (Bt = V_l^T)
  gemm(xh,  V0t, D0, nullptr, nullptr, 0, 1.f, 0.f, nullptr, 0.f,
       nullptr, 0.f, nullptr, 0.f, s1, s1h, Bsz, D1);
  gemm(s1h, V1t, D1, nullptr, nullptr, 0, 1.f, 0.f, nullptr, 0.f,
       nullptr, 0.f, nullptr, 0.f, s2, s2h, Bsz, D1);
  gemm(s2h, V2t, D1, nullptr, nullptr, 0, 1.f, 0.f, nullptr, 0.f,
       nullptr, 0.f, nullptr, 0.f, s3, s3h, Bsz, D0);

  const float n_x   = (float)Bsz * (float)D0;  // numel(x) = numel(s3)
  const float n_mid = (float)Bsz * (float)D1;  // numel(s1) = numel(s2)
  const float cx = 0.2f / n_x;                 // LR*2/numel
  const float cm = 0.2f / n_mid;

  for (int it = 0; it < 10; ++it) {
    // residuals: u_l = s_l@V_l - s_{l+1} (Bt=V_l^T);  d_l = s_{l+1}@W_l - s_l (Bt=W_l^T)
    gemm(xh,  V0t, D0, nullptr, nullptr, 0, 1.f, 0.f, s1, -1.f,
         nullptr, 0.f, nullptr, 0.f, nullptr, u0, Bsz, D1);
    gemm(s1h, V1t, D1, nullptr, nullptr, 0, 1.f, 0.f, s2, -1.f,
         nullptr, 0.f, nullptr, 0.f, nullptr, u1, Bsz, D1);
    gemm(s2h, V2t, D1, nullptr, nullptr, 0, 1.f, 0.f, s3, -1.f,
         nullptr, 0.f, nullptr, 0.f, nullptr, u2, Bsz, D0);
    gemm(s1h, W0t, D1, nullptr, nullptr, 0, 1.f, 0.f, x,  -1.f,
         nullptr, 0.f, nullptr, 0.f, nullptr, d0, Bsz, D0);
    gemm(s2h, W1t, D1, nullptr, nullptr, 0, 1.f, 0.f, s1, -1.f,
         nullptr, 0.f, nullptr, 0.f, nullptr, d1, Bsz, D1);
    gemm(s3h, W2t, D0, nullptr, nullptr, 0, 1.f, 0.f, s2, -1.f,
         nullptr, 0.f, nullptr, 0.f, nullptr, d2, Bsz, D1);

    // fused dual-GEMM updates: s_k = s_k - LR*grad_k   (X@W^T uses Bt=W itself)
    gemm(d0, W0h, D0, u1, V1h, D1, -cx, -cm, s1, 1.f,
         d1,  cm, u0,  cm, s1, s1h, Bsz, D1);
    gemm(d1, W1h, D1, u2, V2h, D0, -cm, -cx, s2, 1.f,
         d2,  cm, u1,  cm, s2, s2h, Bsz, D1);
    gemm(d2, W2h, D1, nullptr, nullptr, 0, -cm, 0.f, s3, 1.f,
         u2,  cx, nullptr, 0.f, s3, s3h, Bsz, D0);
  }

  (void)hipMemcpyAsync(d_out, s3, (size_t)Bsz * D0 * sizeof(float),
                       hipMemcpyDeviceToDevice, stream);
}
